// CrossModelAttention_43173011260023
// MI455X (gfx1250) — compile-verified, matrix operand order checked by reading, not run on hardware
//
#include <hip/hip_runtime.h>

// ---------------------------------------------------------------------------
// CrossModelAttention for MI455X (gfx1250, wave32, WMMA).
// All heavy GEMMs via v_wmma_f32_16x16x32_bf16 (bf16 in, fp32 accumulate).
// Double-buffered LDS pipeline: global loads for tile k+1 overlap WMMAs of k.
// ---------------------------------------------------------------------------

typedef __bf16 bf16_t;
typedef __attribute__((ext_vector_type(16))) __bf16 v16bf;
typedef __attribute__((ext_vector_type(8)))  __bf16 v8bf;
typedef __attribute__((ext_vector_type(8)))  float  v8f;
typedef __attribute__((ext_vector_type(4)))  float  v4f;

#define CB  16      // batch
#define CL  2048    // seq len
#define CR  64      // rank
#define CD  1024    // hidden
#define CDA 512     // d_attn

#define LDS_K 40    // 32 + 8 pad elements -> 80B row stride (16B aligned, conflict-free)

// Load a 16x32 bf16 WMMA operand fragment (A or B side) for this lane.
// layout (ISA 7.12.2, 16-bit A 16x32): lanes 0-15 hold K {kb..kb+7, kb+16..kb+23}
// with kb=0; lanes 16-31 with kb=8. `base` must point at row_ptr + k0 + kb.
__device__ inline v16bf frag_ld(const bf16_t* base) {
    v8bf lo = *reinterpret_cast<const v8bf*>(base);
    v8bf hi = *reinterpret_cast<const v8bf*>(base + 16);
    return __builtin_shufflevector(lo, hi, 0, 1, 2, 3, 4, 5, 6, 7,
                                           8, 9, 10, 11, 12, 13, 14, 15);
}

__device__ inline v8bf pack8(v4f a0, v4f a1) {
    v8bf r;
#pragma unroll
    for (int i = 0; i < 4; ++i) { r[i] = (bf16_t)a0[i]; r[4 + i] = (bf16_t)a1[i]; }
    return r;
}

// ---------------------------------------------------------------------------
// Generic C = A @ W^T + bias, bf16 output.  (template ABF: A dtype bf16/fp32)
//   A: row-major, leading dim lda;  W: fp32 (N x K) row-major
//   cmode 1: row-major bf16 store:  Cb[m*ldc + col_off + n]
//   cmode 2: per-64-row-group transposed bf16 store (for basis_v (l,d,r)):
//            Cb[(m>>6)*(N*64) + n*64 + (m&63)]  — 16B v8bf per lane
// Block: 256 threads (8 waves), tile 128(M) x 128(N), K-step 32.
// Wave (wm,wn) = (wave&1, wave>>1) owns 64x32 -> acc[4][2] v8f.
// ---------------------------------------------------------------------------
template <int ABF>
__global__ void __launch_bounds__(256)
gemm_bf16_wmma(const void* __restrict__ Av, long lda,
               const float* __restrict__ W, const float* __restrict__ bias,
               bf16_t* __restrict__ Cb, long ldc, long col_off, int cmode,
               int M, int N, int K)
{
    __shared__ bf16_t sA[2][128 * LDS_K];
    __shared__ bf16_t sB[2][128 * LDS_K];

    const int tid  = threadIdx.x;
    const int lane = tid & 31;
    const int wave = tid >> 5;
    const int wm   = wave & 1;    // 0..1  (M half)
    const int wn   = wave >> 1;   // 0..3  (N quarter)
    const long m0  = (long)blockIdx.y * 128;
    const long n0  = (long)blockIdx.x * 128;

    v8f acc[4][2];
#pragma unroll
    for (int mt = 0; mt < 4; ++mt)
#pragma unroll
        for (int nt = 0; nt < 2; ++nt)
            acc[mt][nt] = (v8f){0.f, 0.f, 0.f, 0.f, 0.f, 0.f, 0.f, 0.f};

    const int row  = tid >> 1;    // 0..127 staging row
    const int half = tid & 1;     // which 16-element K half
    const int soff = row * LDS_K + half * 16;

    // register staging for the in-flight tile
    v8bf rab[2];                  // A as bf16 (ABF path)
    v4f  raf[4];                  // A as fp32
    v4f  rwf[4];                  // W as fp32

    auto load_regs = [&](int kt) {
        if (ABF) {
            const bf16_t* A = (const bf16_t*)Av + (m0 + row) * lda + kt + half * 16;
            rab[0] = *reinterpret_cast<const v8bf*>(A);
            rab[1] = *reinterpret_cast<const v8bf*>(A + 8);
        } else {
            const float* A = (const float*)Av + (m0 + row) * lda + kt + half * 16;
#pragma unroll
            for (int i = 0; i < 4; ++i) raf[i] = *reinterpret_cast<const v4f*>(A + 4 * i);
        }
        const float* Wp = W + (n0 + row) * (long)K + kt + half * 16;
#pragma unroll
        for (int i = 0; i < 4; ++i) rwf[i] = *reinterpret_cast<const v4f*>(Wp + 4 * i);
    };

    auto store_lds = [&](int buf) {
        if (ABF) {
            *reinterpret_cast<v8bf*>(&sA[buf][soff])     = rab[0];
            *reinterpret_cast<v8bf*>(&sA[buf][soff + 8]) = rab[1];
        } else {
            *reinterpret_cast<v8bf*>(&sA[buf][soff])     = pack8(raf[0], raf[1]);
            *reinterpret_cast<v8bf*>(&sA[buf][soff + 8]) = pack8(raf[2], raf[3]);
        }
        *reinterpret_cast<v8bf*>(&sB[buf][soff])     = pack8(rwf[0], rwf[1]);
        *reinterpret_cast<v8bf*>(&sB[buf][soff + 8]) = pack8(rwf[2], rwf[3]);
    };

    const int kb = (lane < 16) ? 0 : 8;
    const int ln = lane & 15;

    // ---- pipelined main loop ----
    load_regs(0);
    store_lds(0);
    __syncthreads();

    const int iters = K / 32;
    for (int it = 0; it < iters; ++it) {
        const bool has_next = (it + 1) < iters;
        if (has_next) load_regs((it + 1) * 32);    // global loads overlap WMMAs below

        const bf16_t* bufA = sA[it & 1];
        const bf16_t* bufB = sB[it & 1];
        v16bf afrag[4], bfrag[2];
#pragma unroll
        for (int mt = 0; mt < 4; ++mt)
            afrag[mt] = frag_ld(&bufA[(wm * 64 + mt * 16 + ln) * LDS_K + kb]);
#pragma unroll
        for (int nt = 0; nt < 2; ++nt)
            bfrag[nt] = frag_ld(&bufB[(wn * 32 + nt * 16 + ln) * LDS_K + kb]);

#pragma unroll
        for (int mt = 0; mt < 4; ++mt)
#pragma unroll
            for (int nt = 0; nt < 2; ++nt)
                acc[mt][nt] = __builtin_amdgcn_wmma_f32_16x16x32_bf16(
                    false, afrag[mt], false, bfrag[nt],
                    (short)0, acc[mt][nt], false, false);

        if (has_next) store_lds((it + 1) & 1);     // safe: readers of that buffer
        __syncthreads();                            // finished before prev barrier
    }

    // ---- bias + store ----
    const int lh = (lane >> 4) & 1;
#pragma unroll
    for (int nt = 0; nt < 2; ++nt) {
        const long n  = n0 + wn * 32 + nt * 16 + ln;
        const float bn = bias ? bias[n] : 0.f;
#pragma unroll
        for (int mt = 0; mt < 4; ++mt) {
            const long mbase = m0 + wm * 64 + mt * 16 + lh * 8;
            v8f a = acc[mt][nt];
            if (cmode == 2) {
                v8bf pk;
#pragma unroll
                for (int v = 0; v < 8; ++v) pk[v] = (bf16_t)(a[v] + bn);
                const long grp = mbase >> 6;
                const long r   = mbase & 63;
                *reinterpret_cast<v8bf*>(&Cb[grp * ((long)N * 64) + n * 64 + r]) = pk;
            } else {
#pragma unroll
                for (int v = 0; v < 8; ++v)
                    Cb[(mbase + v) * ldc + col_off + n] = (bf16_t)(a[v] + bn);
            }
        }
    }
}

// ---------------------------------------------------------------------------
// Per-position attention: one block per l (128 threads = 4 waves).
//   scores = qk_l (16x512) @ bk_l^T (512x64) * scale ; softmax over 64
//   z      = w (16x64) @ bv_l (64x512)  -> out fp32
// qk: bf16 (B,L,DA); bk: bf16 (L,R,DA); bvT: bf16 (L,DA,R)
// ---------------------------------------------------------------------------
#define QK_STR 520   // 512 + 8 pad
#define BK_STR 520
#define W_STR  72    // 64 + 8 pad

__global__ void __launch_bounds__(128)
attn_wmma(const bf16_t* __restrict__ qk, const bf16_t* __restrict__ bk,
          const bf16_t* __restrict__ bvT, float* __restrict__ out)
{
    __shared__ bf16_t sQK[16 * QK_STR];
    __shared__ bf16_t sBK[64 * BK_STR];
    __shared__ bf16_t sW[4 * 16 * W_STR];

    const int l    = blockIdx.x;
    const int tid  = threadIdx.x;
    const int lane = tid & 31;
    const int wave = tid >> 5;

    // ---- stage qk_l: 16 rows (batch) x 512, gathered across batch stride ----
#pragma unroll
    for (int i = 0; i < 8; ++i) {
        const int c = i * 128 + tid;          // 0..1023 chunks of 8 bf16
        const int b = c >> 6;
        const int col = (c & 63) * 8;
        *reinterpret_cast<v8bf*>(&sQK[b * QK_STR + col]) =
            *reinterpret_cast<const v8bf*>(qk + ((long)b * CL + l) * CDA + col);
    }
    // ---- stage bk_l: 64 x 512 ----
    const bf16_t* bkl = bk + (long)l * CR * CDA;
#pragma unroll
    for (int i = 0; i < 32; ++i) {
        const int c = i * 128 + tid;          // 0..4095
        const int r = c >> 6;
        const int col = (c & 63) * 8;
        *reinterpret_cast<v8bf*>(&sBK[r * BK_STR + col]) =
            *reinterpret_cast<const v8bf*>(bkl + r * CDA + col);
    }
    __syncthreads();

    const int kb = (lane < 16) ? 0 : 8;
    const int ln = lane & 15;
    const int lh = (lane >> 4) & 1;

    // ---- scores: every wave computes the full 16x64 tile (K = 512) ----
    v8f sc[4];
#pragma unroll
    for (int t = 0; t < 4; ++t) sc[t] = (v8f){0.f, 0.f, 0.f, 0.f, 0.f, 0.f, 0.f, 0.f};

    for (int kt = 0; kt < CDA; kt += 32) {
        const v16bf a = frag_ld(&sQK[ln * QK_STR + kt + kb]);
#pragma unroll
        for (int t = 0; t < 4; ++t) {
            const v16bf bf = frag_ld(&sBK[(t * 16 + ln) * BK_STR + kt + kb]);
            sc[t] = __builtin_amdgcn_wmma_f32_16x16x32_bf16(
                false, a, false, bf, (short)0, sc[t], false, false);
        }
    }

    const float scale = 0.04419417382415922f;   // 1/sqrt(512)
#pragma unroll
    for (int t = 0; t < 4; ++t)
#pragma unroll
        for (int v = 0; v < 8; ++v) sc[t][v] *= scale;

    // ---- softmax over R=64 per row (rows live across 16-lane groups) ----
    v8f mx;
#pragma unroll
    for (int v = 0; v < 8; ++v) {
        float m = sc[0][v];
#pragma unroll
        for (int t = 1; t < 4; ++t) m = fmaxf(m, sc[t][v]);
        mx[v] = m;
    }
#pragma unroll
    for (int off = 1; off < 16; off <<= 1)
#pragma unroll
        for (int v = 0; v < 8; ++v)
            mx[v] = fmaxf(mx[v], __shfl_xor((float)mx[v], off, 32));

    v8f sum = (v8f){0.f, 0.f, 0.f, 0.f, 0.f, 0.f, 0.f, 0.f};
#pragma unroll
    for (int t = 0; t < 4; ++t)
#pragma unroll
        for (int v = 0; v < 8; ++v) {
            sc[t][v] = __expf(sc[t][v] - mx[v]);
            sum[v] += sc[t][v];
        }
#pragma unroll
    for (int off = 1; off < 16; off <<= 1)
#pragma unroll
        for (int v = 0; v < 8; ++v)
            sum[v] += __shfl_xor((float)sum[v], off, 32);

    // ---- re-layout weights (C-layout -> A-fragment) via per-wave LDS ----
    bf16_t* w = &sW[wave * 16 * W_STR];
#pragma unroll
    for (int t = 0; t < 4; ++t)
#pragma unroll
        for (int v = 0; v < 8; ++v) {
            const int m = v + lh * 8;
            w[m * W_STR + t * 16 + ln] = (bf16_t)(sc[t][v] / sum[v]);
        }
    __syncthreads();

    v16bf af[2];
#pragma unroll
    for (int k2 = 0; k2 < 2; ++k2)
        af[k2] = frag_ld(&w[ln * W_STR + k2 * 32 + kb]);

    // ---- z: wave owns d range [wave*128, wave*128+128), B from global (l,d,r) ----
    const bf16_t* bvl = bvT + (long)l * CDA * CR;
#pragma unroll
    for (int dt = 0; dt < 8; ++dt) {
        const int n0 = wave * 128 + dt * 16;
        v8f az = (v8f){0.f, 0.f, 0.f, 0.f, 0.f, 0.f, 0.f, 0.f};
#pragma unroll
        for (int k2 = 0; k2 < 2; ++k2) {
            const bf16_t* p = bvl + (n0 + ln) * CR + k2 * 32 + kb;
            v8bf lo = *reinterpret_cast<const v8bf*>(p);
            v8bf hi = *reinterpret_cast<const v8bf*>(p + 16);
            v16bf bf = __builtin_shufflevector(lo, hi, 0, 1, 2, 3, 4, 5, 6, 7,
                                                       8, 9, 10, 11, 12, 13, 14, 15);
            az = __builtin_amdgcn_wmma_f32_16x16x32_bf16(
                false, af[k2], false, bf, (short)0, az, false, false);
        }
#pragma unroll
        for (int v = 0; v < 8; ++v) {
            const int b = v + lh * 8;
            out[((long)b * CL + l) * CDA + n0 + ln] = az[v];
        }
    }
}

// ---------------------------------------------------------------------------
extern "C" void kernel_launch(void* const* d_in, const int* in_sizes, int n_in,
                              void* d_out, int out_size, void* d_ws, size_t ws_size,
                              hipStream_t stream)
{
    const float* student = (const float*)d_in[0];   // (B,L,D)
    const float* teacher = (const float*)d_in[1];   // (B,L,D)
    const float* basis   = (const float*)d_in[2];   // (L,R,D)
    const float* q_W  = (const float*)d_in[3];
    const float* q_b  = (const float*)d_in[4];
    const float* k_W  = (const float*)d_in[5];
    const float* k_b  = (const float*)d_in[6];
    const float* qk_W = (const float*)d_in[7];      // (DA, 2*DA)
    const float* qk_b = (const float*)d_in[8];
    const float* bk_W = (const float*)d_in[9];
    const float* bk_b = (const float*)d_in[10];
    const float* bv_W = (const float*)d_in[11];
    const float* bv_b = (const float*)d_in[12];
    float* out = (float*)d_out;

    // workspace carve (bf16): [q|k] 64MB, qk 32MB, basis_k 128MB, basis_v^T 128MB
    bf16_t* qkin = (bf16_t*)d_ws;                              // (32768, 1024)
    bf16_t* qkv  = qkin + (size_t)32768 * 1024;                // (32768, 512)
    bf16_t* bkv  = qkv  + (size_t)32768 * 512;                 // (L, R, DA)
    bf16_t* bvT  = bkv  + (size_t)131072 * 512;                // (L, DA, R)

    const dim3 blk(256);
    const int ML = CB * CL;      // 32768
    const int MB = CL * CR;      // 131072

    // q -> qkin[:, 0:512]
    gemm_bf16_wmma<0><<<dim3(4, ML / 128), blk, 0, stream>>>(
        student, CD, q_W, q_b, qkin, 2 * CDA, 0, 1, ML, CDA, CD);
    // k -> qkin[:, 512:1024]
    gemm_bf16_wmma<0><<<dim3(4, ML / 128), blk, 0, stream>>>(
        teacher, CD, k_W, k_b, qkin, 2 * CDA, CDA, 1, ML, CDA, CD);
    // qk = [q|k] @ qk_W^T  (bf16 A path)
    gemm_bf16_wmma<1><<<dim3(4, ML / 128), blk, 0, stream>>>(
        qkin, 2 * CDA, qk_W, qk_b, qkv, CDA, 0, 1, ML, CDA, 2 * CDA);
    // basis_k (row-major (l,r,d))
    gemm_bf16_wmma<0><<<dim3(4, MB / 128), blk, 0, stream>>>(
        basis, CD, bk_W, bk_b, bkv, CDA, 0, 1, MB, CDA, CD);
    // basis_v (group-transposed -> (l,d,r))
    gemm_bf16_wmma<0><<<dim3(4, MB / 128), blk, 0, stream>>>(
        basis, CD, bv_W, bv_b, bvT, 0, 0, 2, MB, CDA, CD);
    // attention + softmax + z
    attn_wmma<<<dim3(CL), dim3(128), 0, stream>>>(qkv, bkv, bvT, out);
}
